// TransformerBlock_24343874633890
// MI455X (gfx1250) — compile-verified, matrix operand order checked by reading, not run on hardware
//
#include <hip/hip_runtime.h>

// ---------------------------------------------------------------------------
// MI455X (gfx1250) transformer block: fp32 I/O, bf16 WMMA compute.
// Compute-bound (~240 GFLOP vs ~130MB HBM) -> everything through
// v_wmma_f32_16x16x32_bf16 with f32 accumulation. GEMM staging uses
// global_load_async_to_lds_b128 (ASYNCcnt) with LDS double buffering.
// ---------------------------------------------------------------------------

typedef __attribute__((ext_vector_type(16))) __bf16 bf16x16;
typedef __attribute__((ext_vector_type(8)))  float  f32x8;

#define DIMC  1024
#define INTER 4096
#define NTOK  4096   // B * S
#define SEQ   2048
#define NHEAD 16
#define HDIM  64

// round-to-nearest-even f32 -> bf16
__device__ __forceinline__ __bf16 f2bf(float f) {
  unsigned u = __builtin_bit_cast(unsigned, f);
  u += 0x7FFFu + ((u >> 16) & 1u);
  unsigned short s = (unsigned short)(u >> 16);
  return __builtin_bit_cast(__bf16, s);
}

// Async global->LDS copy of 16 bytes (tracked by ASYNCcnt).
// LDS operand is the workgroup-relative byte offset: generic shared-aperture
// flat addresses keep the LDS offset in addr[31:0], so truncation is exact.
__device__ __forceinline__ void async_g2l_b128(void* lds, const void* gaddr) {
  const unsigned loff = (unsigned)(unsigned long long)lds;
  asm volatile("global_load_async_to_lds_b128 %0, %1, off"
               :: "v"(loff), "v"(gaddr) : "memory");
}

template <int N>
__device__ __forceinline__ void wait_asynccnt() {
  asm volatile("s_wait_asynccnt %0" :: "i"(N) : "memory");
}

// Load one 16x32 bf16 WMMA A/B fragment from a row-major [16][ldr] region.
// ISA layout: lanes 0-15 hold row=lane, K in [0,8)+[16,24);
//             lanes 16-31 hold row=lane-16, K in [8,16)+[24,32).
__device__ __forceinline__ bf16x16 frag16x32(const __bf16* p, int ldr) {
  const int lane = threadIdx.x & 31;
  const int r  = lane & 15;
  const int ko = (lane >> 4) << 3;
  union { bf16x16 v; uint4 q[2]; } u;
  u.q[0] = *(const uint4*)(p + (size_t)r * ldr + ko);
  u.q[1] = *(const uint4*)(p + (size_t)r * ldr + ko + 16);
  return u.v;
}

__device__ __forceinline__ f32x8 wmma_bf16(bf16x16 a, bf16x16 b, f32x8 c) {
  return __builtin_amdgcn_wmma_f32_16x16x32_bf16(false, a, false, b,
                                                 (short)0, c, false, false);
}

// ---------------------------------------------------------------------------
// Weight cast+transpose: W[K][N] f32 -> Wt[N][K] bf16. 32x32 tiles via LDS.
// ---------------------------------------------------------------------------
__global__ __launch_bounds__(256)
void transpose_cast(const float* __restrict__ W, __bf16* __restrict__ Wt,
                    int K, int N) {
  __shared__ __bf16 tile[32][33];
  const int bn = blockIdx.x * 32, bk = blockIdx.y * 32;
  const int tx = threadIdx.x & 31, ty = threadIdx.x >> 5;   // ty: 0..7
  #pragma unroll
  for (int i = 0; i < 32; i += 8)
    tile[ty + i][tx] = f2bf(W[(size_t)(bk + ty + i) * N + bn + tx]);
  __syncthreads();
  #pragma unroll
  for (int i = 0; i < 32; i += 8)
    Wt[(size_t)(bn + ty + i) * K + bk + tx] = tile[tx][ty + i];
}

// ---------------------------------------------------------------------------
// RMSNorm + cast to bf16. One row (1024 f32) per 256-thread block.
// ---------------------------------------------------------------------------
__global__ __launch_bounds__(256)
void rmsnorm_cast(const float* __restrict__ X, const float* __restrict__ W,
                  __bf16* __restrict__ Y) {
  const int row = blockIdx.x;
  const int t = threadIdx.x;
  const float4 v = ((const float4*)(X + (size_t)row * DIMC))[t];
  float ss = v.x * v.x + v.y * v.y + v.z * v.z + v.w * v.w;
  #pragma unroll
  for (int off = 16; off; off >>= 1) ss += __shfl_xor(ss, off);
  __shared__ float red[8];
  if ((t & 31) == 0) red[t >> 5] = ss;
  __syncthreads();
  float tot = 0.f;
  #pragma unroll
  for (int i = 0; i < 8; ++i) tot += red[i];
  const float sc = rsqrtf(tot * (1.0f / DIMC) + 1e-6f);
  const float4 w = ((const float4*)W)[t];
  __bf16* y = Y + (size_t)row * DIMC + t * 4;
  y[0] = f2bf(v.x * sc * w.x);
  y[1] = f2bf(v.y * sc * w.y);
  y[2] = f2bf(v.z * sc * w.z);
  y[3] = f2bf(v.w * sc * w.w);
}

// ---------------------------------------------------------------------------
// WMMA GEMM: C[M][N] = A[M][K](bf16) x Bt[N][K](bf16)^T
// 128x128 block tile, BK=32, 8 waves in 2(M)x4(N), 64x32 per wave.
// Async global->LDS staging, double buffered (copy overlaps WMMA).
// EPI 0: bf16 store | 1: f32 + residual | 2: dual-B fused SwiGLU, bf16 store
// ---------------------------------------------------------------------------
template <int EPI>
__global__ __launch_bounds__(256)
void gemm_bf16(const __bf16* __restrict__ A, const __bf16* __restrict__ Bt,
               const __bf16* __restrict__ Bt2, const float* __restrict__ R,
               float* __restrict__ Cf, __bf16* __restrict__ Cb,
               int M, int N, int K) {
  __shared__ __bf16 sA [2][128 * 40];   // pad 40: 80B rows, 16B aligned
  __shared__ __bf16 sB [2][128 * 40];
  __shared__ __bf16 sB2[2][128 * 40];

  const int mBase = blockIdx.y * 128;
  const int nBase = blockIdx.x * 128;
  const int wave = threadIdx.x >> 5;
  const int lane = threadIdx.x & 31;
  const int wM = (wave & 1) * 64;
  const int wN = (wave >> 1) * 32;

  // per-thread async ops issued per tile (in-order completion per wave)
  constexpr int NOPS = (EPI == 2) ? 6 : 4;

  const auto stage = [&](int buf, int k0) {
    #pragma unroll
    for (int i = 0; i < 2; ++i) {
      const int idx = threadIdx.x * 2 + i;          // 0..511
      const int row = idx >> 2, ch = (idx & 3) << 3;
      async_g2l_b128(sA[buf] + row * 40 + ch,
                     A + (size_t)(mBase + row) * K + k0 + ch);
      async_g2l_b128(sB[buf] + row * 40 + ch,
                     Bt + (size_t)(nBase + row) * K + k0 + ch);
      if constexpr (EPI == 2)
        async_g2l_b128(sB2[buf] + row * 40 + ch,
                       Bt2 + (size_t)(nBase + row) * K + k0 + ch);
    }
  };

  f32x8 acc [4][2] = {};
  f32x8 acc2[4][2] = {};

  stage(0, 0);
  int pb = 0;
  for (int k0 = 0; k0 < K; k0 += 32, pb ^= 1) {
    const bool more = (k0 + 32 < K);
    if (more) {
      stage(pb ^ 1, k0 + 32);          // overlap next copy with this compute
      wait_asynccnt<NOPS>();           // tile k0 done; only next tile in flight
    } else {
      wait_asynccnt<0>();
    }
    __syncthreads();                    // all waves' tile-k0 LDS writes visible

    bf16x16 af[4];
    #pragma unroll
    for (int mt = 0; mt < 4; ++mt)
      af[mt] = frag16x32(sA[pb] + (wM + mt * 16) * 40, 40);
    #pragma unroll
    for (int nt = 0; nt < 2; ++nt) {
      bf16x16 bfr = frag16x32(sB[pb] + (wN + nt * 16) * 40, 40);
      #pragma unroll
      for (int mt = 0; mt < 4; ++mt)
        acc[mt][nt] = wmma_bf16(af[mt], bfr, acc[mt][nt]);
      if constexpr (EPI == 2) {
        bf16x16 b2 = frag16x32(sB2[pb] + (wN + nt * 16) * 40, 40);
        #pragma unroll
        for (int mt = 0; mt < 4; ++mt)
          acc2[mt][nt] = wmma_bf16(af[mt], b2, acc2[mt][nt]);
      }
    }
    __syncthreads();                    // done reading before buffer reuse
  }

  // C layout: elem r of accum -> row = r + 8*(lane>=16), col = lane&15
  const int rhalf = (lane >> 4) * 8;
  #pragma unroll
  for (int mt = 0; mt < 4; ++mt)
    #pragma unroll
    for (int nt = 0; nt < 2; ++nt)
      #pragma unroll
      for (int r = 0; r < 8; ++r) {
        const int row = mBase + wM + mt * 16 + r + rhalf;
        const int col = nBase + wN + nt * 16 + (lane & 15);
        const size_t off = (size_t)row * N + col;
        const float v = acc[mt][nt][r];
        if constexpr (EPI == 0) {
          Cb[off] = f2bf(v);
        } else if constexpr (EPI == 1) {
          Cf[off] = v + R[off];
        } else {
          const float g = acc2[mt][nt][r];
          const float s = v / (1.0f + __expf(-v));   // silu
          Cb[off] = f2bf(s * g);
        }
      }
}

// ---------------------------------------------------------------------------
// Flash attention (causal). Grid: (S/64 q-tiles, B*H). 128 threads = 4 waves,
// 16 q-rows per wave, KV tiles of 32, online softmax in f32.
// Mask only applied on each wave's diagonal tile; fully-masked tiles skip
// compute (barriers + cooperative V staging remain uniform).
// ---------------------------------------------------------------------------
__global__ __launch_bounds__(128)
void attn_kernel(const __bf16* __restrict__ Q, const __bf16* __restrict__ Kb,
                 const __bf16* __restrict__ Vb, __bf16* __restrict__ O) {
  __shared__ __bf16 sVT[64 * 40];      // V^T: [d=64][k=32], pad 40
  __shared__ __bf16 sP [4 * 16 * 40];  // per-wave P: [16 q][32 k], pad 40

  const int qt = blockIdx.x;
  const int b  = blockIdx.y >> 4;
  const int h  = blockIdx.y & 15;
  const int wv = threadIdx.x >> 5;
  const int lane = threadIdx.x & 31;
  const int rl = lane & 15, rh = lane >> 4;

  const size_t headoff = (size_t)b * SEQ * DIMC + (size_t)h * HDIM;
  const __bf16* Qp = Q  + headoff;
  const __bf16* Kp = Kb + headoff;
  const __bf16* Vp = Vb + headoff;
  __bf16*       Op = O  + headoff;

  const int qbase = qt * 64;
  const int qrow  = qbase + wv * 16;   // wave's first q row

  bf16x16 qf[2];
  #pragma unroll
  for (int dd = 0; dd < 2; ++dd)
    qf[dd] = frag16x32(Qp + (size_t)qrow * DIMC + dd * 32, DIMC);

  float m[8], l[8];
  #pragma unroll
  for (int r = 0; r < 8; ++r) { m[r] = -3.0e38f; l[r] = 0.f; }
  f32x8 o[4] = {};

  for (int kb = 0; kb < qbase + 64; kb += 32) {
    // cooperative V^T staging: 32 k-rows x 64 d (always uniform)
    {
      const int k  = threadIdx.x >> 2;
      const int dc = (threadIdx.x & 3) * 16;
      union { uint4 q[2]; __bf16 e[16]; } u;
      const __bf16* vr = Vp + (size_t)(kb + k) * DIMC + dc;
      u.q[0] = *(const uint4*)(vr);
      u.q[1] = *(const uint4*)(vr + 8);
      #pragma unroll
      for (int j = 0; j < 16; ++j) sVT[(dc + j) * 40 + k] = u.e[j];
    }

    const bool active = (kb <= qrow + 15);        // tile intersects causal region
    const bool diag   = active && (kb + 31 > qrow); // tile crosses the diagonal

    if (active) {
      // scores: two 16x16 subtiles; K fragments straight from global
      f32x8 sc[2];
      #pragma unroll
      for (int sub = 0; sub < 2; ++sub) {
        f32x8 a = {};
        #pragma unroll
        for (int dd = 0; dd < 2; ++dd) {
          bf16x16 kf =
              frag16x32(Kp + (size_t)(kb + sub * 16) * DIMC + dd * 32, DIMC);
          a = wmma_bf16(qf[dd], kf, a);
        }
        a = a * 0.125f;                 // 1/sqrt(64)
        if (diag) {
          #pragma unroll
          for (int r = 0; r < 8; ++r) {
            const int qg = qrow + r + rh * 8;
            const int kg = kb + sub * 16 + rl;
            if (kg > qg) a[r] = -3.0e38f;
          }
        }
        sc[sub] = a;
      }

      // online softmax; row reductions stay inside 16-lane halves
      #pragma unroll
      for (int r = 0; r < 8; ++r) {
        float mx = fmaxf(sc[0][r], sc[1][r]);
        #pragma unroll
        for (int off = 1; off < 16; off <<= 1)
          mx = fmaxf(mx, __shfl_xor(mx, off));
        const float mn = fmaxf(m[r], mx);
        const float alpha = __expf(m[r] - mn);
        const float p0 = __expf(sc[0][r] - mn);
        const float p1 = __expf(sc[1][r] - mn);
        float ps = p0 + p1;
        #pragma unroll
        for (int off = 1; off < 16; off <<= 1) ps += __shfl_xor(ps, off);
        l[r] = l[r] * alpha + ps;
        m[r] = mn;
        #pragma unroll
        for (int ds = 0; ds < 4; ++ds) o[ds][r] *= alpha;
        const int prow = r + rh * 8;
        sP[wv * 640 + prow * 40 + rl]      = f2bf(p0);
        sP[wv * 640 + prow * 40 + 16 + rl] = f2bf(p1);
      }
    }

    __syncthreads();   // P + V^T visible

    if (active) {
      bf16x16 pf = frag16x32(sP + wv * 640, 40);
      #pragma unroll
      for (int ds = 0; ds < 4; ++ds) {
        bf16x16 vf = frag16x32(sVT + ds * 16 * 40, 40);
        o[ds] = wmma_bf16(pf, vf, o[ds]);
      }
    }
    __syncthreads();   // done reading LDS before next tile overwrites
  }

  #pragma unroll
  for (int r = 0; r < 8; ++r) {
    const float inv = 1.0f / l[r];
    const int row = qrow + r + rh * 8;
    #pragma unroll
    for (int ds = 0; ds < 4; ++ds)
      Op[(size_t)row * DIMC + ds * 16 + rl] = f2bf(o[ds][r] * inv);
  }
}

// ---------------------------------------------------------------------------
// Launch: weight prep -> ln1 -> QKV -> attn -> O+res -> ln2 -> SwiGLU -> down+res
// Workspace (88 MB): [0,32M) weights bf16 | [32,40M) norm buf | [40,56M) h f32
// [56,88M) q/k/v/attn bf16, later aliased by the 32MB SwiGLU activation.
// ---------------------------------------------------------------------------
extern "C" void kernel_launch(void* const* d_in, const int* in_sizes, int n_in,
                              void* d_out, int out_size, void* d_ws, size_t ws_size,
                              hipStream_t stream) {
  const float* x    = (const float*)d_in[0];
  const float* ln1w = (const float*)d_in[1];
  const float* wq   = (const float*)d_in[2];
  const float* wk   = (const float*)d_in[3];
  const float* wv   = (const float*)d_in[4];
  const float* wo   = (const float*)d_in[5];
  const float* ln2w = (const float*)d_in[6];
  const float* w1   = (const float*)d_in[7];
  const float* w3   = (const float*)d_in[8];
  const float* w2   = (const float*)d_in[9];
  float* out = (float*)d_out;

  char* ws = (char*)d_ws;
  const size_t MB = 1024u * 1024u;
  __bf16* wq_t = (__bf16*)(ws + 0);
  __bf16* wk_t = (__bf16*)(ws + 2 * MB);
  __bf16* wv_t = (__bf16*)(ws + 4 * MB);
  __bf16* wo_t = (__bf16*)(ws + 6 * MB);
  __bf16* w1_t = (__bf16*)(ws + 8 * MB);
  __bf16* w3_t = (__bf16*)(ws + 16 * MB);
  __bf16* w2_t = (__bf16*)(ws + 24 * MB);
  __bf16* xn   = (__bf16*)(ws + 32 * MB);     // ln1 out, reused for ln2 out
  float*  h    = (float*)(ws + 40 * MB);      // attn residual sum, f32
  __bf16* qb   = (__bf16*)(ws + 56 * MB);
  __bf16* kb   = (__bf16*)(ws + 64 * MB);
  __bf16* vb   = (__bf16*)(ws + 72 * MB);
  __bf16* ab   = (__bf16*)(ws + 80 * MB);
  __bf16* gate = (__bf16*)(ws + 56 * MB);     // aliases q/k/v/attn (dead by then)

  const dim3 blk256(256);

  // weights -> bf16 [N][K]
  transpose_cast<<<dim3(DIMC/32,  DIMC/32 ), blk256, 0, stream>>>(wq, wq_t, DIMC,  DIMC);
  transpose_cast<<<dim3(DIMC/32,  DIMC/32 ), blk256, 0, stream>>>(wk, wk_t, DIMC,  DIMC);
  transpose_cast<<<dim3(DIMC/32,  DIMC/32 ), blk256, 0, stream>>>(wv, wv_t, DIMC,  DIMC);
  transpose_cast<<<dim3(DIMC/32,  DIMC/32 ), blk256, 0, stream>>>(wo, wo_t, DIMC,  DIMC);
  transpose_cast<<<dim3(INTER/32, DIMC/32 ), blk256, 0, stream>>>(w1, w1_t, DIMC,  INTER);
  transpose_cast<<<dim3(INTER/32, DIMC/32 ), blk256, 0, stream>>>(w3, w3_t, DIMC,  INTER);
  transpose_cast<<<dim3(DIMC/32,  INTER/32), blk256, 0, stream>>>(w2, w2_t, INTER, DIMC);

  // ln1
  rmsnorm_cast<<<NTOK, blk256, 0, stream>>>(x, ln1w, xn);

  // QKV projections
  gemm_bf16<0><<<dim3(DIMC/128, NTOK/128), blk256, 0, stream>>>(
      xn, wq_t, nullptr, nullptr, nullptr, qb, NTOK, DIMC, DIMC);
  gemm_bf16<0><<<dim3(DIMC/128, NTOK/128), blk256, 0, stream>>>(
      xn, wk_t, nullptr, nullptr, nullptr, kb, NTOK, DIMC, DIMC);
  gemm_bf16<0><<<dim3(DIMC/128, NTOK/128), blk256, 0, stream>>>(
      xn, wv_t, nullptr, nullptr, nullptr, vb, NTOK, DIMC, DIMC);

  // causal flash attention
  attn_kernel<<<dim3(SEQ/64, 2 * NHEAD), dim3(128), 0, stream>>>(qb, kb, vb, ab);

  // O projection + residual -> h
  gemm_bf16<1><<<dim3(DIMC/128, NTOK/128), blk256, 0, stream>>>(
      ab, wo_t, nullptr, x, h, nullptr, NTOK, DIMC, DIMC);

  // ln2
  rmsnorm_cast<<<NTOK, blk256, 0, stream>>>(h, ln2w, xn);

  // fused SwiGLU up-projections: silu(n@w1) * (n@w3) -> bf16
  gemm_bf16<2><<<dim3(INTER/128, NTOK/128), blk256, 0, stream>>>(
      xn, w1_t, w3_t, nullptr, nullptr, gate, NTOK, INTER, DIMC);

  // down projection + residual -> out
  gemm_bf16<1><<<dim3(DIMC/128, NTOK/128), blk256, 0, stream>>>(
      gate, w2_t, nullptr, h, out, nullptr, NTOK, DIMC, INTER);

  (void)in_sizes; (void)n_in; (void)out_size; (void)ws_size;
}
